// RQVAE_60928406061060
// MI455X (gfx1250) — matrix operand order repaired and checked by reading, not compiled, not run on hardware
//
#include <hip/hip_runtime.h>
#include <hip/hip_bf16.h>

// ---------------------------------------------------------------------------
// RQ-VAE forward for MI455X (gfx1250, wave32, WMMA).
// All GEMMs run through a 128x128-tile bf16 WMMA kernel (f32 accumulate),
// double-buffered LDS, natural-K-order staging so each WMMA operand is two
// contiguous ds_load_b128's. f32->bf16 staging packs via one v_perm_b32.
// The 8192x8192 distance matrices are never materialized: the distance GEMM
// fuses an argmin epilogue.
// ---------------------------------------------------------------------------

typedef __attribute__((ext_vector_type(16))) __bf16 v16bf;
typedef __attribute__((ext_vector_type(8)))  __bf16 v8bf;
typedef __attribute__((ext_vector_type(8)))  float  v8f;

// pack two f32 -> two bf16 (round-to-nearest, ties up) in one dword:
// two v_add + one v_perm_b32 (result.lo16 = bf16(a), result.hi16 = bf16(b))
__device__ __forceinline__ unsigned pk2bf(float a, float b) {
    union { float f; unsigned u; } x, y; x.f = a; y.f = b;
    return __builtin_amdgcn_perm(y.u + 0x8000u, x.u + 0x8000u, 0x07060302u);
}

__device__ __forceinline__ v16bf ldfrag(const __bf16* p) {
    const v8bf lo = *(const v8bf*)p;          // ds_load_b128
    const v8bf hi = *(const v8bf*)(p + 8);    // ds_load_b128
    return __builtin_shufflevector(lo, hi, 0, 1, 2, 3, 4, 5, 6, 7,
                                   8, 9, 10, 11, 12, 13, 14, 15);
}

#define TM  128
#define TN  128
#define TK  32
#define TKP 40   // LDS row stride in bf16 elems: 80 B, 16B-aligned, bank-spread

// A-tile K-block permutation so each half-wave's fragment is contiguous:
// blocks [0,1,2,3] (8 K each) -> positions [0,2,1,3].
__device__ __forceinline__ int permA(int c4) {
    const int blk = c4 >> 3;
    const int nb  = (blk == 1) ? 2 : ((blk == 2) ? 1 : blk);
    return (nb << 3) | (c4 & 7);
}

// MODE: 0 = bias+ReLU, 1 = bias only, 2 = argmin distance epilogue (no C write)
// TRANSB: false -> B is [K,N] row-major (weights); true -> B is [N,K] row-major
template <int MODE, bool TRANSB>
__global__ __launch_bounds__(256)
void gemm_wmma_bf16(const float* __restrict__ A, const float* __restrict__ B,
                    const float* __restrict__ bias, float* __restrict__ C,
                    int M, int N, int K,
                    const float* __restrict__ cnorm, float* __restrict__ partial) {
    __shared__ __bf16 sA[2][TM][TKP];   // A tiles (natural K order)
    __shared__ __bf16 sB[2][TN][TKP];   // B^T tiles (natural K order)
    __shared__ float  redv[TM][4];
    __shared__ int    redi[TM][4];

    const int tid  = threadIdx.x;
    const int lane = tid & 31;
    const int wave = tid >> 5;            // 0..7
    const int wm   = wave >> 2;           // 0..1  (64-row strip)
    const int wn   = wave & 3;            // 0..3  (32-col strip)
    const int half = lane >> 4;
    const int l16  = lane & 15;

    const int m0 = blockIdx.y * TM;
    const int n0 = blockIdx.x * TN;

    // staging registers
    float4 ra[4];
    float4 rbv[4];     // TRANSB path
    float  rbs[16];    // weight path

    const int arow = (tid >> 3);          // 0..31 (base, +32 per pass)
    const int ac4  = (tid & 7) * 4;
    const int bn   = tid & 127;           // weight path: column
    const int bko  = (tid >> 7) * 2;      // weight path: k sub-offset

    auto loadT = [&](int k0) {
        #pragma unroll
        for (int p = 0; p < 4; ++p)
            ra[p] = *(const float4*)(A + (size_t)(m0 + p * 32 + arow) * K + k0 + ac4);
        if constexpr (TRANSB) {
            #pragma unroll
            for (int p = 0; p < 4; ++p)
                rbv[p] = *(const float4*)(B + (size_t)(n0 + p * 32 + arow) * K + k0 + ac4);
        } else {
            #pragma unroll
            for (int p = 0; p < 8; ++p) {
                const int kk = p * 4 + bko;
                rbs[2 * p]     = B[(size_t)(k0 + kk)     * N + n0 + bn];
                rbs[2 * p + 1] = B[(size_t)(k0 + kk + 1) * N + n0 + bn];
            }
        }
    };
    auto storeT = [&](int buf) {
        const int apos = ac4;
        #pragma unroll
        for (int p = 0; p < 4; ++p) {
            uint2 q;
            q.x = pk2bf(ra[p].x, ra[p].y);
            q.y = pk2bf(ra[p].z, ra[p].w);
            *(uint2*)&sA[buf][p * 32 + arow][apos] = q;     // ds_store_b64
        }
        if constexpr (TRANSB) {
            #pragma unroll
            for (int p = 0; p < 4; ++p) {
                uint2 q;
                q.x = pk2bf(rbv[p].x, rbv[p].y);
                q.y = pk2bf(rbv[p].z, rbv[p].w);
                *(uint2*)&sB[buf][p * 32 + arow][ac4] = q;  // natural K order
            }
        } else {
            #pragma unroll
            for (int p = 0; p < 8; ++p) {
                const int kk = p * 4 + bko;
                *(unsigned*)&sB[buf][bn][kk] = pk2bf(rbs[2 * p], rbs[2 * p + 1]);
            }
        }
    };

    v8f acc[4][2] = {};

    loadT(0);
    storeT(0);
    __syncthreads();

    const int nk = K / TK;
    for (int t = 0; t < nk; ++t) {
        const int  cur  = t & 1;
        const bool more = (t + 1 < nk);

        if (more) loadT((t + 1) * TK);                 // overlap with WMMA below
        if (t + 2 < nk) {                              // near-temporal prefetch
            const int kp = (t + 2) * TK;
            __builtin_prefetch(A + (size_t)(m0 + (tid >> 1)) * K + kp + (tid & 1) * 16, 0, 3);
            if (TRANSB)
                __builtin_prefetch(B + (size_t)(n0 + (tid >> 1)) * K + kp + (tid & 1) * 16, 0, 3);
            else
                __builtin_prefetch(B + (size_t)(kp + (tid >> 3)) * N + n0 + (tid & 7) * 16, 0, 3);
        }

        v16bf af[4], bfr[2];
        #pragma unroll
        for (int mi = 0; mi < 4; ++mi)
            af[mi] = ldfrag(&sA[cur][wm * 64 + mi * 16 + l16][half * 16]);
        #pragma unroll
        for (int ni = 0; ni < 2; ++ni)
            bfr[ni] = ldfrag(&sB[cur][wn * 32 + ni * 16 + l16][half * 16]);

        #pragma unroll
        for (int mi = 0; mi < 4; ++mi)
            #pragma unroll
            for (int ni = 0; ni < 2; ++ni)
                acc[mi][ni] = __builtin_amdgcn_wmma_f32_16x16x32_bf16(
                    false, af[mi], false, bfr[ni], (short)0, acc[mi][ni], false, false);

        if (more) {
            storeT(cur ^ 1);
            __syncthreads();
        }
    }

    if (MODE != 2) {
        // C/D layout: lane l16 = column; VGPR v = row v + 8*half within subtile.
        #pragma unroll
        for (int mi = 0; mi < 4; ++mi) {
            #pragma unroll
            for (int ni = 0; ni < 2; ++ni) {
                const int col = n0 + wn * 32 + ni * 16 + l16;
                const float bv = bias[col];
                #pragma unroll
                for (int v = 0; v < 8; ++v) {
                    const int row = m0 + wm * 64 + mi * 16 + v + 8 * half;
                    float r = acc[mi][ni][v] + bv;
                    if (MODE == 0) r = fmaxf(r, 0.0f);
                    C[(size_t)row * N + col] = r;
                }
            }
        }
    } else {
        // argmin over this block's 128 columns of dist = cnorm[k] - 2*(r.e_k)
        const int nStrips = gridDim.x;
        #pragma unroll
        for (int mi = 0; mi < 4; ++mi) {
            #pragma unroll
            for (int v = 0; v < 8; ++v) {
                const int col0 = n0 + wn * 32 + l16;
                float val = cnorm[col0] - 2.0f * acc[mi][0][v];
                int   idx = col0;
                {
                    const float d1 = cnorm[col0 + 16] - 2.0f * acc[mi][1][v];
                    if (d1 < val || (d1 == val && col0 + 16 < idx)) { val = d1; idx = col0 + 16; }
                }
                #pragma unroll
                for (int off = 1; off < 16; off <<= 1) {   // per-half-wave butterfly
                    const float ov = __shfl_xor(val, off, 16);
                    const int   oi = __shfl_xor(idx, off, 16);
                    if (ov < val || (ov == val && oi < idx)) { val = ov; idx = oi; }
                }
                if (l16 == 0) {
                    const int rl = wm * 64 + mi * 16 + v + 8 * half;
                    redv[rl][wn] = val;
                    redi[rl][wn] = idx;
                }
            }
        }
        __syncthreads();
        if (tid < TM) {
            float bv = redv[tid][0];
            int   bi = redi[tid][0];
            #pragma unroll
            for (int w = 1; w < 4; ++w) {
                const float v = redv[tid][w];
                const int   i = redi[tid][w];
                if (v < bv || (v == bv && i < bi)) { bv = v; bi = i; }
            }
            const size_t base = ((size_t)(m0 + tid) * nStrips + blockIdx.x) * 2;
            partial[base] = bv;
            ((int*)partial)[base + 1] = bi;
        }
    }
}

// ---------------------------------------------------------------------------
__global__ void reparam_kernel(const float* __restrict__ mu, const float* __restrict__ lv,
                               const float* __restrict__ eps, float* __restrict__ resid,
                               float* __restrict__ qsum, int n) {
    const int i = blockIdx.x * blockDim.x + threadIdx.x;
    if (i < n) {
        const float z = mu[i] + eps[i] * __expf(0.5f * lv[i]);
        resid[i] = z;
        qsum[i]  = 0.0f;
    }
}

__global__ void codebook_norms(const float* __restrict__ E, float* __restrict__ cnorm, int D) {
    const int k = blockIdx.x;
    const int tid = threadIdx.x;              // 256 threads, D == 256
    float v = E[(size_t)k * D + tid];
    v *= v;
    #pragma unroll
    for (int off = 16; off > 0; off >>= 1) v += __shfl_xor(v, off, 32);
    __shared__ float s[8];
    if ((tid & 31) == 0) s[tid >> 5] = v;
    __syncthreads();
    if (tid == 0) {
        float t = 0.0f;
        #pragma unroll
        for (int i = 0; i < 8; ++i) t += s[i];
        cnorm[k] = t;
    }
}

__global__ void rvq_finalize(const float* __restrict__ partial, int nStrips,
                             const float* __restrict__ E, float* __restrict__ qsum,
                             float* __restrict__ resid, float* __restrict__ codes,
                             int lvl, int D) {
    const int row = blockIdx.x;
    const int tid = threadIdx.x;              // 256 threads
    __shared__ float sv[64];
    __shared__ int   si[64];
    if (tid < nStrips) {
        const size_t base = ((size_t)row * nStrips + tid) * 2;
        sv[tid] = partial[base];
        si[tid] = ((const int*)partial)[base + 1];
    }
    __syncthreads();
    for (int s = 32; s > 0; s >>= 1) {
        if (tid < s && tid + s < nStrips) {
            if (sv[tid + s] < sv[tid] || (sv[tid + s] == sv[tid] && si[tid + s] < si[tid])) {
                sv[tid] = sv[tid + s];
                si[tid] = si[tid + s];
            }
        }
        __syncthreads();
    }
    const int c = si[0];
    const float q = E[(size_t)c * D + tid];
    qsum [(size_t)row * D + tid] += q;
    resid[(size_t)row * D + tid] -= q;
    if (tid == 0) codes[row * 4 + lvl] = (float)c;
}

// ---------------------------------------------------------------------------
extern "C" void kernel_launch(void* const* d_in, const int* in_sizes, int n_in,
                              void* d_out, int out_size, void* d_ws, size_t ws_size,
                              hipStream_t stream) {
    constexpr int    Bsz  = 8192;   // batch
    constexpr int    Din  = 1024;
    constexpr int    H    = 2048;
    constexpr int    Dz   = 256;
    constexpr int    KCB  = 8192;   // codebook size
    constexpr int    NLVL = 4;

    const float* x      = (const float*)d_in[0];
    const float* eps    = (const float*)d_in[1];
    const float* enc_w1 = (const float*)d_in[2];
    const float* enc_b1 = (const float*)d_in[3];
    const float* enc_w2 = (const float*)d_in[4];
    const float* enc_b2 = (const float*)d_in[5];
    const float* mu_w   = (const float*)d_in[6];
    const float* mu_b   = (const float*)d_in[7];
    const float* lv_w   = (const float*)d_in[8];
    const float* lv_b   = (const float*)d_in[9];
    const float* cbooks = (const float*)d_in[10];
    const float* dec_w1 = (const float*)d_in[11];
    const float* dec_b1 = (const float*)d_in[12];
    const float* dec_w2 = (const float*)d_in[13];
    const float* dec_b2 = (const float*)d_in[14];
    const float* dec_w3 = (const float*)d_in[15];
    const float* dec_b3 = (const float*)d_in[16];

    // ---- output layout (return order, flat) ----
    float* out    = (float*)d_out;
    float* recon  = out;                                   // [8192,1024]
    float* out_mu = recon  + (size_t)Bsz * Din;            // [8192,256]
    float* out_lv = out_mu + (size_t)Bsz * Dz;             // [8192,256]
    float* qsum   = out_lv + (size_t)Bsz * Dz;             // [8192,256]
    float* codes  = qsum   + (size_t)Bsz * Dz;             // [8192,4] (as float)

    // ---- workspace layout ----
    char* ws = (char*)d_ws;
    constexpr size_t SZ_H1 = (size_t)Bsz * H   * 4;        // 64 MB
    constexpr size_t SZ_H2 = (size_t)Bsz * Din * 4;        // 32 MB
    constexpr size_t SZ_RZ = (size_t)Bsz * Dz  * 4;        //  8 MB
    constexpr size_t SZ_CN = (size_t)KCB * 4;
    float* h1      = (float*)(ws);
    float* h2      = (float*)(ws + SZ_H1);
    float* resid   = (float*)(ws + SZ_H1 + SZ_H2);
    float* cnorm   = (float*)(ws + SZ_H1 + SZ_H2 + SZ_RZ);
    float* partial = (float*)(ws + SZ_H1 + SZ_H2 + SZ_RZ + SZ_CN);  // [8192][64] * {f32,i32}

    const dim3 blk(256);

    // ---- encoder ----
    gemm_wmma_bf16<0, false><<<dim3(H   / TN, Bsz / TM), blk, 0, stream>>>(
        x,  enc_w1, enc_b1, h1, Bsz, H,   Din, nullptr, nullptr);
    gemm_wmma_bf16<0, false><<<dim3(Din / TN, Bsz / TM), blk, 0, stream>>>(
        h1, enc_w2, enc_b2, h2, Bsz, Din, H,   nullptr, nullptr);
    gemm_wmma_bf16<1, false><<<dim3(Dz  / TN, Bsz / TM), blk, 0, stream>>>(
        h2, mu_w,   mu_b,   out_mu, Bsz, Dz, Din, nullptr, nullptr);
    gemm_wmma_bf16<1, false><<<dim3(Dz  / TN, Bsz / TM), blk, 0, stream>>>(
        h2, lv_w,   lv_b,   out_lv, Bsz, Dz, Din, nullptr, nullptr);

    // ---- reparameterize; init residual=z, qsum=0 ----
    {
        const int n = Bsz * Dz;
        reparam_kernel<<<(n + 255) / 256, blk, 0, stream>>>(out_mu, out_lv, eps,
                                                            resid, qsum, n);
    }

    // ---- residual vector quantization (4 levels) ----
    for (int lvl = 0; lvl < NLVL; ++lvl) {
        const float* E = cbooks + (size_t)lvl * KCB * Dz;
        codebook_norms<<<KCB, blk, 0, stream>>>(E, cnorm, Dz);
        gemm_wmma_bf16<2, true><<<dim3(KCB / TN, Bsz / TM), blk, 0, stream>>>(
            resid, E, nullptr, nullptr, Bsz, KCB, Dz, cnorm, partial);
        rvq_finalize<<<Bsz, blk, 0, stream>>>(partial, KCB / TN, E, qsum, resid,
                                              codes, lvl, Dz);
    }

    // ---- decoder ----
    gemm_wmma_bf16<0, false><<<dim3(Din / TN, Bsz / TM), blk, 0, stream>>>(
        qsum, dec_w1, dec_b1, h2, Bsz, Din, Dz,  nullptr, nullptr);
    gemm_wmma_bf16<0, false><<<dim3(H   / TN, Bsz / TM), blk, 0, stream>>>(
        h2,   dec_w2, dec_b2, h1, Bsz, H,   Din, nullptr, nullptr);
    gemm_wmma_bf16<1, false><<<dim3(Din / TN, Bsz / TM), blk, 0, stream>>>(
        h1,   dec_w3, dec_b3, recon, Bsz, Din, H, nullptr, nullptr);
}
